// CenterLoss_39891656245569
// MI455X (gfx1250) — matrix operand order repaired and checked by reading, not compiled, not run on hardware
//
#include <hip/hip_runtime.h>

typedef __attribute__((ext_vector_type(16))) _Float16 v16h;
typedef __attribute__((ext_vector_type(8)))  _Float16 v8h;
typedef __attribute__((ext_vector_type(8)))  float    v8f;
typedef __attribute__((ext_vector_type(4)))  float    v4f;

#define B_N      131072
#define D_N      512
#define N_C      101
#define NPAD     112            // 7 tiles of 16
#define BIGV     1e9f
#define EPS_N    0.1f
#define CBLOCKS  1024
#define CTHREADS 256

// ---------------------------------------------------------------------------
// Kernel 1: per-sample squared distance to own center, block partial sums.
// 268 MB single-use embedding stream > 192 MB L2  ->  non-temporal loads
// (TH=NT) so the stream does not evict the hot 207 KB centers table.
// ---------------------------------------------------------------------------
__global__ void center_partial_kernel(const float* __restrict__ emb,
                                      const float* __restrict__ centers,
                                      const int*   __restrict__ labels,
                                      float*       __restrict__ partials) {
    __shared__ float red[CTHREADS];
    const int tid  = threadIdx.x;
    const int wave = tid >> 5;
    const int lane = tid & 31;
    // 128 samples per block, 16 samples per wave (8 waves)
    const int sample0 = blockIdx.x * 128 + wave * 16;

    float acc = 0.f;
    for (int s = 0; s < 16; ++s) {
        const int i = sample0 + s;
        const int c = labels[i];
        const v4f* e = (const v4f*)(emb     + (size_t)i * D_N);
        const v4f* w = (const v4f*)(centers + (size_t)c * D_N);
#pragma unroll
        for (int it = 0; it < 4; ++it) {
            v4f ev = __builtin_nontemporal_load(&e[lane + it * 32]); // streaming
            v4f wv = w[lane + it * 32];                              // keep cached
            v4f d  = ev - wv;
            acc += d.x * d.x + d.y * d.y + d.z * d.z + d.w * d.w;
        }
    }
    red[tid] = acc;
    __syncthreads();
    for (int off = CTHREADS / 2; off > 0; off >>= 1) {
        if (tid < off) red[tid] += red[tid + off];
        __syncthreads();
    }
    if (tid == 0) partials[blockIdx.x] = red[0];
}

// ---------------------------------------------------------------------------
// Kernel 2: F.normalize(W, p=2, eps=0.1) -> f16, zero-pad rows to NPAD.
// ---------------------------------------------------------------------------
__global__ void normalize_kernel(const float* __restrict__ centers,
                                 _Float16*    __restrict__ Ch) {
    __shared__ float red[128];
    __shared__ float inv_s;
    const int row = blockIdx.x;    // 0..111
    const int tid = threadIdx.x;   // 128

    if (row >= N_C) {
        for (int c = tid; c < D_N; c += 128) Ch[(size_t)row * D_N + c] = (_Float16)0.f;
        return;
    }
    const float* w = centers + (size_t)row * D_N;
    float s = 0.f;
    for (int c = tid; c < D_N; c += 128) { float v = w[c]; s += v * v; }
    red[tid] = s;
    __syncthreads();
    for (int off = 64; off > 0; off >>= 1) {
        if (tid < off) red[tid] += red[tid + off];
        __syncthreads();
    }
    if (tid == 0) inv_s = 1.f / fmaxf(sqrtf(red[0]), EPS_N);
    __syncthreads();
    const float inv = inv_s;
    for (int c = tid; c < D_N; c += 128)
        Ch[(size_t)row * D_N + c] = (_Float16)(w[c] * inv);
}

// ---------------------------------------------------------------------------
// Kernel 3: Gram = Ch * Ch^T via v_wmma_f32_16x16x32_f16; Dist = 1 - Gram.
// One wave per 16x16 output tile (EXEC all-ones as WMMA requires).
// A fragment (16x32 f16): lane m=L%16; 16 contiguous K values,
// lanes>=16 add +16.  B fragment (32x16): lane n=L%16,
// 16 contiguous K values, lanes>=16 add +16.  Both are row reads of Ch.
// ---------------------------------------------------------------------------
__global__ void gram_kernel(const _Float16* __restrict__ Ch,
                            float*          __restrict__ Dist) {
    const int lane = threadIdx.x;       // 32 threads = 1 wave
    const int tile = blockIdx.x;        // 0..48
    const int rt = tile / 7, ct = tile % 7;
    const int m  = lane & 15;
    const bool hi = lane >= 16;

    const _Float16* arow = Ch + (size_t)(rt * 16 + m) * D_N + (hi ? 16 : 0);
    const _Float16* brow = Ch + (size_t)(ct * 16 + m) * D_N + (hi ? 16 : 0);

    v8f c = {};
    for (int kk = 0; kk < D_N; kk += 32) {
        v8h alo = *(const v8h*)(arow + kk);        // K = base .. base+7
        v8h ahi = *(const v8h*)(arow + kk + 8);    // K = base+8 .. base+15
        v16h a, b;
#pragma unroll
        for (int h = 0; h < 8; ++h) { a[h] = alo[h]; a[h + 8] = ahi[h]; }
        b = *(const v16h*)(brow + kk);             // 16 contiguous K values
        c = __builtin_amdgcn_wmma_f32_16x16x32_f16(
                /*neg_a=*/false, a, /*neg_b=*/false, b,
                /*c_mod=*/(short)0, c, /*reuse_a=*/false, /*reuse_b=*/false);
    }

    // C/D layout: VGPR r -> row (r + 8*hi), col = lane%16
    const int col   = ct * 16 + m;
    const int rbase = rt * 16 + (hi ? 8 : 0);
#pragma unroll
    for (int r = 0; r < 8; ++r)
        Dist[(rbase + r) * NPAD + col] = 1.0f - c[r];
}

// ---------------------------------------------------------------------------
// Kernel 4: masked row-min, neighbor ReLU margin, combine with center mean.
// ---------------------------------------------------------------------------
__global__ void final_kernel(const float* __restrict__ partials,
                             const float* __restrict__ Dist,
                             float*       __restrict__ out) {
    __shared__ float redm[128];
    __shared__ float redc[128];
    const int tid = threadIdx.x;

    float contrib = 0.f;
    if (tid < N_C) {
        const int i = tid;
        float minv = 1e30f;
        for (int j = 0; j < N_C; ++j) {
            float d = Dist[i * NPAD + j];
            int diff = i - j; if (diff < 0) diff = -diff;
            float dm = d + (diff == 0 ? BIGV : 0.f) + (diff == 1 ? BIGV : 0.f);
            minv = fminf(minv, dm);
        }
        if (i >= 1) {                                   // d_lower path
            float dl = Dist[i * NPAD + (i - 1)];
            contrib += fmaxf(2.f * dl - minv, 0.f);
        }
        if (i <= N_C - 2) {                             // d_upper path
            float du = Dist[i * NPAD + (i + 1)];
            contrib += fmaxf(2.f * du - minv, 0.f);
        }
    }
    float cpart = 0.f;
    for (int k = tid; k < CBLOCKS; k += 128) cpart += partials[k];

    redm[tid] = contrib;
    redc[tid] = cpart;
    __syncthreads();
    for (int off = 64; off > 0; off >>= 1) {
        if (tid < off) { redm[tid] += redm[tid + off]; redc[tid] += redc[tid + off]; }
        __syncthreads();
    }
    if (tid == 0)
        out[0] = redm[0] / (float)N_C + redc[0] / (float)B_N;
}

// ---------------------------------------------------------------------------
extern "C" void kernel_launch(void* const* d_in, const int* in_sizes, int n_in,
                              void* d_out, int out_size, void* d_ws, size_t ws_size,
                              hipStream_t stream) {
    const float* emb     = (const float*)d_in[0];
    const float* centers = (const float*)d_in[1];
    const int*   labels  = (const int*)d_in[2];
    float*       out     = (float*)d_out;

    // workspace layout
    float*    partials = (float*)d_ws;                                   // 1024 f32
    _Float16* Ch       = (_Float16*)((char*)d_ws + 4096);                // 112*512 f16
    float*    Dist     = (float*)((char*)d_ws + 4096 + NPAD * D_N * 2);  // 112*112 f32

    center_partial_kernel<<<CBLOCKS, CTHREADS, 0, stream>>>(emb, centers, labels, partials);
    normalize_kernel<<<NPAD, 128, 0, stream>>>(centers, Ch);
    gram_kernel<<<49, 32, 0, stream>>>(Ch, Dist);
    final_kernel<<<1, 128, 0, stream>>>(partials, Dist, out);
}
